// LSTM_19104014533029
// MI455X (gfx1250) — matrix operand order repaired and checked by reading, not compiled, not run on hardware
//
#include <hip/hip_runtime.h>

typedef __attribute__((ext_vector_type(16))) __bf16 v16bf;
typedef __attribute__((ext_vector_type(2)))  __bf16 bf16x2;
typedef __attribute__((ext_vector_type(8)))  float  v8f;

#define BTILE 16
#define HDIM  128
#define HP16W 68       // padded row stride of packed-bf16 h rows, in dwords (64 data + 4 pad)
#define TT    60
#define NG    4

union BFrag { v16bf v; unsigned u[8]; };

// hardware float->bf16 conversion, packed pair in one dword (v_cvt_pk_bf16_f32)
__device__ __forceinline__ unsigned pack2_bf16(float lo, float hi) {
  bf16x2 t;
  t.x = (__bf16)lo;
  t.y = (__bf16)hi;
  return __builtin_bit_cast(unsigned, t);
}

__device__ __forceinline__ float fsigmoid(float x) {
  return 1.0f / (1.0f + __expf(-x));
}
__device__ __forceinline__ float ftanh_fast(float x) {
  float e = __expf(2.0f * x);
  return 1.0f - 2.0f / (e + 1.0f);   // saturates correctly at +-1
}

// CDNA5 split workgroup barrier for the read->write hazard on h in LDS.
// Signal only after our LDS reads have completed (s_wait_dscnt 0), then run the
// long transcendental chain, then wait: barrier skew overlaps with compute.
__device__ __forceinline__ void wg_barrier_signal_after_reads() {
  asm volatile("s_wait_dscnt 0\n\ts_barrier_signal -1" ::: "memory");
}
__device__ __forceinline__ void wg_barrier_wait() {
  asm volatile("s_barrier_wait -1" ::: "memory");
}

// Load 4 gate-tiles x 4 K-chunk B-fragments of a row-major [512,128] weight.
// B-frag (16-bit, 32x16): VGPR j<4: K=8*lh+2j,+1; VGPR j>=4: K=16+8*lh+2(j-4),+1; N=lane%16.
// gates[m][n] = sum_k h[m][k]*w[n][k]  =>  B[k][n] = w[n][k]
__device__ __forceinline__ void load_wB(BFrag bw[NG][4], const float* __restrict__ w,
                                        int wv, int lm, int lh) {
#pragma unroll
  for (int g = 0; g < NG; ++g) {
#pragma unroll
    for (int kc = 0; kc < 4; ++kc) {
      const float* p = w + (size_t)(g * HDIM + wv * 16 + lm) * HDIM + kc * 32 + lh * 8;
#pragma unroll
      for (int j = 0; j < 8; ++j)
        bw[g][kc].u[j] = pack2_bf16(p[2 * j + (j >= 4 ? 8 : 0)], p[2 * j + 1 + (j >= 4 ? 8 : 0)]);
    }
  }
}

// A-frag (16-bit, 16x32): lane row M = lane%16.
// VGPR j<4: K = 32kc + 8lh + 2j(,+1);  VGPR j>=4: K = 32kc + 8lh + 16 + 2(j-4)(,+1)
// With h packed as bf16 pairs per dword: a.u[j] = h16row[16kc + 4lh + j (+8 for j>=4)]
// -> two aligned ds_load_b128 per K-chunk, zero conversion ALU.
__device__ __forceinline__ void build_A16(BFrag a[4], const unsigned* h16row, int lh) {
#pragma unroll
  for (int kc = 0; kc < 4; ++kc) {
    const int d0 = kc * 16 + lh * 4;
#pragma unroll
    for (int j = 0; j < 4; ++j) a[kc].u[j]     = h16row[d0 + j];
#pragma unroll
    for (int j = 0; j < 4; ++j) a[kc].u[4 + j] = h16row[d0 + 8 + j];
  }
}

#define WMMA_BF16(A, B_, C_) \
  __builtin_amdgcn_wmma_f32_16x16x32_bf16(false, (A), false, (B_), (short)0, (C_), false, false)

__global__ __launch_bounds__(256) void lstm2_wmma_kernel(
    const float* __restrict__ noise,   // [B,60,4]
    const float* __restrict__ w_ih1,   // [512,4]
    const float* __restrict__ w_hh1,   // [512,128]
    const float* __restrict__ b_ih1,
    const float* __restrict__ b_hh1,
    const float* __restrict__ w_ih2,   // [512,128]
    const float* __restrict__ w_hh2,   // [512,128]
    const float* __restrict__ b_ih2,
    const float* __restrict__ b_hh2,
    const float* __restrict__ fc_w,    // [128]
    const float* __restrict__ fc_b,    // [1]
    float* __restrict__ out,           // [B,60]
    int Bn)
{
  __shared__ __align__(16) unsigned h16[BTILE][HP16W];  // h as packed bf16 pairs
  __shared__ float x_lds[BTILE][TT * 4];
  __shared__ float fcw_lds[HDIM];
  __shared__ float fc_part[BTILE][9];

  const int tid  = threadIdx.x;
  const int wv   = tid >> 5;        // wave 0..7 -> owns hidden cols 16*wv .. 16*wv+15
  const int lane = tid & 31;
  const int lm   = lane & 15;
  const int lh   = lane >> 4;
  const int m0   = blockIdx.x * BTILE;

  // Stage x tile (whole sequence for these 16 rows); zero h16.
  for (int i = tid; i < BTILE * TT * 4; i += 256) {
    int r = i / (TT * 4), cc = i % (TT * 4);
    x_lds[r][cc] = (m0 + r < Bn) ? noise[(size_t)(m0 + r) * (TT * 4) + cc] : 0.f;
  }
  for (int i = tid; i < BTILE * HP16W; i += 256) h16[i / HP16W][i % HP16W] = 0u;
  if (tid < HDIM) fcw_lds[tid] = fc_w[tid];

  // Per-lane gate column (same hidden col across all 4 gates -> lane-local cell update).
  const int col = wv * 16 + lm;
  float wih[NG][4], bias1[NG], bias2[NG];
#pragma unroll
  for (int g = 0; g < NG; ++g) {
    int gc = g * HDIM + col;
#pragma unroll
    for (int k = 0; k < 4; ++k) wih[g][k] = w_ih1[gc * 4 + k];
    bias1[g] = b_ih1[gc] + b_hh1[gc];
    bias2[g] = b_ih2[gc] + b_hh2[gc];
  }
  const float fcb = fc_b[0];

  BFrag bw[NG][4];                       // 16 B-frags pinned in VGPRs across the loop
  load_wB(bw, w_hh1, wv, lm, lh);

  float c8[8];
#pragma unroll
  for (int v = 0; v < 8; ++v) c8[v] = 0.f;

  const v8f vzero = {0.f, 0.f, 0.f, 0.f, 0.f, 0.f, 0.f, 0.f};

  // per-lane bf16 store slots: row r = 8*lh+v, element index col
  __bf16* const hb_base = (__bf16*)&h16[0][0];

  __syncthreads();

  // ================= layer 1 =================
  for (int t = 0; t < TT; ++t) {
    BFrag a[4];
    build_A16(a, &h16[lm][0], lh);
    v8f acc[NG];
#pragma unroll
    for (int g = 0; g < NG; ++g) acc[g] = vzero;
#pragma unroll
    for (int kc = 0; kc < 4; ++kc)
#pragma unroll
      for (int g = 0; g < NG; ++g)
        acc[g] = WMMA_BF16(a[kc].v, bw[g][kc].v, acc[g]);
    // our h_{t-1} loads have completed -> safe to signal; wait after the VALU chain
    wg_barrier_signal_after_reads();
    float hv8[8];
#pragma unroll
    for (int v = 0; v < 8; ++v) {
      const int r = lh * 8 + v;         // C layout: VGPR v -> row 8*lh+v, col = lane%16
      float gi = acc[0][v] + bias1[0];
      float gf = acc[1][v] + bias1[1];
      float gg = acc[2][v] + bias1[2];
      float go = acc[3][v] + bias1[3];
#pragma unroll
      for (int k = 0; k < 4; ++k) {     // I=4 input contribution, on the fly
        float xv = x_lds[r][t * 4 + k];
        gi += xv * wih[0][k];
        gf += xv * wih[1][k];
        gg += xv * wih[2][k];
        go += xv * wih[3][k];
      }
      float cv = fsigmoid(gf) * c8[v] + fsigmoid(gi) * ftanh_fast(gg);
      c8[v] = cv;
      hv8[v] = fsigmoid(go) * ftanh_fast(cv);
    }
    wg_barrier_wait();                  // all waves' reads of h_{t-1} done
#pragma unroll
    for (int v = 0; v < 8; ++v)
      hb_base[(lh * 8 + v) * (2 * HP16W) + col] = (__bf16)hv8[v];  // ds_store_b16
    __syncthreads();                    // h_t visible before next A build
  }

  // ===== layer-2 constant input gates: xg2 = h1_last @ w_ih2^T + b (computed once) =====
  load_wB(bw, w_ih2, wv, lm, lh);
  v8f xg2[NG];
  {
    BFrag a[4];
    build_A16(a, &h16[lm][0], lh);
#pragma unroll
    for (int g = 0; g < NG; ++g) xg2[g] = vzero;
#pragma unroll
    for (int kc = 0; kc < 4; ++kc)
#pragma unroll
      for (int g = 0; g < NG; ++g)
        xg2[g] = WMMA_BF16(a[kc].v, bw[g][kc].v, xg2[g]);
#pragma unroll
    for (int g = 0; g < NG; ++g)
#pragma unroll
      for (int v = 0; v < 8; ++v) xg2[g][v] += bias2[g];
  }
  load_wB(bw, w_hh2, wv, lm, lh);

  __syncthreads();                      // everyone done reading h1_last
  for (int i = tid; i < BTILE * HP16W; i += 256) h16[i / HP16W][i % HP16W] = 0u;
#pragma unroll
  for (int v = 0; v < 8; ++v) c8[v] = 0.f;
  __syncthreads();

  // ================= layer 2 + FC head =================
  for (int t = 0; t < TT; ++t) {
    BFrag a[4];
    build_A16(a, &h16[lm][0], lh);
    v8f acc[NG];
#pragma unroll
    for (int g = 0; g < NG; ++g) acc[g] = xg2[g];   // constant input-gate contribution
#pragma unroll
    for (int kc = 0; kc < 4; ++kc)
#pragma unroll
      for (int g = 0; g < NG; ++g)
        acc[g] = WMMA_BF16(a[kc].v, bw[g][kc].v, acc[g]);
    wg_barrier_signal_after_reads();
    float hv8[8];
#pragma unroll
    for (int v = 0; v < 8; ++v) {
      float cv = fsigmoid(acc[1][v]) * c8[v] + fsigmoid(acc[0][v]) * ftanh_fast(acc[2][v]);
      c8[v] = cv;
      hv8[v] = fsigmoid(acc[3][v]) * ftanh_fast(cv);
    }
    wg_barrier_wait();
#pragma unroll
    for (int v = 0; v < 8; ++v)
      hb_base[(lh * 8 + v) * (2 * HP16W) + col] = (__bf16)hv8[v];
    __syncthreads();
    // FC head: out[b,t] = tanh(h2 . fc_w + fc_b); 128 threads do partials, 16 finalize
    if (tid < 128) {
      int r = tid >> 3, s = tid & 7;
      float p = 0.f;
#pragma unroll
      for (int j = 0; j < 8; ++j) {     // 8 packed dwords = 16 hidden cols
        unsigned d = h16[r][s * 8 + j];
        float lo = __uint_as_float(d << 16);
        float hi = __uint_as_float(d & 0xFFFF0000u);
        p += lo * fcw_lds[s * 16 + 2 * j] + hi * fcw_lds[s * 16 + 2 * j + 1];
      }
      fc_part[r][s] = p;
    }
    __syncthreads();
    if (tid < 16 && (m0 + tid) < Bn) {
      float sum = fcb;
#pragma unroll
      for (int s = 0; s < 8; ++s) sum += fc_part[tid][s];
      out[(size_t)(m0 + tid) * TT + t] = ftanh_fast(sum);
    }
    __syncthreads();
  }
}

extern "C" void kernel_launch(void* const* d_in, const int* in_sizes, int n_in,
                              void* d_out, int out_size, void* d_ws, size_t ws_size,
                              hipStream_t stream) {
  (void)n_in; (void)out_size; (void)d_ws; (void)ws_size;
  const float* noise = (const float*)d_in[0];
  const float* w_ih1 = (const float*)d_in[1];
  const float* w_hh1 = (const float*)d_in[2];
  const float* b_ih1 = (const float*)d_in[3];
  const float* b_hh1 = (const float*)d_in[4];
  const float* w_ih2 = (const float*)d_in[5];
  const float* w_hh2 = (const float*)d_in[6];
  const float* b_ih2 = (const float*)d_in[7];
  const float* b_hh2 = (const float*)d_in[8];
  const float* fc_w  = (const float*)d_in[9];
  const float* fc_b  = (const float*)d_in[10];
  float* out = (float*)d_out;

  int Bn = in_sizes[0] / (TT * 4);
  int blocks = (Bn + BTILE - 1) / BTILE;
  lstm2_wmma_kernel<<<blocks, 256, 0, stream>>>(noise, w_ih1, w_hh1, b_ih1, b_hh1,
                                                w_ih2, w_hh2, b_ih2, b_hh2,
                                                fc_w, fc_b, out, Bn);
}
